// POSSM_Backbone_S4D_10926396801048
// MI455X (gfx1250) — matrix operand order repaired and checked by reading, not compiled, not run on hardware
//
#include <hip/hip_runtime.h>
#include <math.h>

typedef __attribute__((ext_vector_type(16))) _Float16 v16h;
typedef __attribute__((ext_vector_type(8)))  float    v8f;
typedef __attribute__((ext_vector_type(4)))  int      v4i;

#define T_LEN 2048
#define H_DIM 512
#define DIN   2048
#define MROWS 16384   // B*T
#define LSTR  40      // LDS row stride (f16) for 32-wide K tiles (16B-pad-friendly)

// ---------------- CDNA5 async-to-LDS plumbing (guarded) ---------------------
#ifndef __has_builtin
#define __has_builtin(x) 0
#endif

#if __has_builtin(__builtin_amdgcn_global_load_async_to_lds_b128)
#define USE_ASYNC 1
__device__ __forceinline__ void async_cp16(const _Float16* g, _Float16* l) {
  __builtin_amdgcn_global_load_async_to_lds_b128(
      (__attribute__((address_space(1))) v4i*)g,
      (__attribute__((address_space(3))) v4i*)l, 0, 0);
}
#if __has_builtin(__builtin_amdgcn_s_wait_asynccnt)
#define WAIT_ASYNC(n) __builtin_amdgcn_s_wait_asynccnt(n)
#else
#define WAIT_ASYNC(n) asm volatile("s_wait_asynccnt %0" ::"n"(n))
#endif
#else
#define USE_ASYNC 0
#define WAIT_ASYNC(n) ((void)0)
#endif

// ---------------- WMMA helpers (16x16x32 f16 -> f32 acc) --------------------

__device__ __forceinline__ v8f wmma_f16(v16h a, v16h b, v8f c) {
  return __builtin_amdgcn_wmma_f32_16x16x32_f16(false, a, false, b, (short)0, c,
                                                false, false);
}

// A fragment: lane holds row m=lane&15; per ISA layout two 16B runs.
__device__ __forceinline__ v16h frag_A(const _Float16* As, int row0, int lane) {
  const int m = lane & 15, half = lane >> 4;
  const _Float16* base = As + (row0 + m) * LSTR;
  union { v16h v; unsigned u[8]; } f;
#pragma unroll
  for (int j = 0; j < 8; ++j) {
    const int kb = (j < 4) ? (8 * half + 2 * j) : (16 + 8 * half + 2 * (j - 4));
    f.u[j] = *(const unsigned*)(base + kb);
  }
  return f.v;
}

// B fragment from LDS staged as BsT[n][k] (column-major w.r.t. the 32x16 tile).
__device__ __forceinline__ v16h frag_B(const _Float16* BsT, int col0, int lane) {
  const int n = lane & 15, half = lane >> 4;
  const _Float16* base = BsT + (col0 + n) * LSTR;
  union { v16h v; unsigned u[8]; } f;
#pragma unroll
  for (int j = 0; j < 8; ++j) {
    const int kb = (j < 4) ? (8 * half + 2 * j) : (16 + 8 * half + 2 * (j - 4));
    f.u[j] = *(const unsigned*)(base + kb);
  }
  return f.v;
}

// Stage one 128-row x 32-col f16 tile (row-major source, ld = K) into LDS.
// 8KB = 512 x 16B: 2 async/vector ops per thread. All addresses 16B aligned.
__device__ __forceinline__ void stage_t(const _Float16* __restrict__ base,
                                        int r0, int k0, int K,
                                        _Float16* lds, int t) {
#pragma unroll
  for (int i = 0; i < 2; ++i) {
    const int idx = t + i * 256;
    const int r = idx >> 2, c8 = (idx & 3) * 8;
    const _Float16* g = base + (size_t)(r0 + r) * K + k0 + c8;
    _Float16* l = lds + r * LSTR + c8;
#if USE_ASYNC
    async_cp16(g, l);
#else
    *(uint4*)(void*)l = *(const uint4*)(const void*)g;
#endif
  }
}

// ---------------- prep: f32->f16 convert / transpose-convert ----------------

__global__ __launch_bounds__(256) void k_cvt(const float* __restrict__ src,
                                             _Float16* __restrict__ dst) {
  const size_t i8 = ((size_t)blockIdx.x * 256 + threadIdx.x) * 8;
  const float4 a = *(const float4*)(src + i8);
  const float4 b = *(const float4*)(src + i8 + 4);
  union { _Float16 h[8]; uint4 u; } o;
  o.h[0] = (_Float16)a.x; o.h[1] = (_Float16)a.y;
  o.h[2] = (_Float16)a.z; o.h[3] = (_Float16)a.w;
  o.h[4] = (_Float16)b.x; o.h[5] = (_Float16)b.y;
  o.h[6] = (_Float16)b.z; o.h[7] = (_Float16)b.w;
  *(uint4*)(dst + i8) = o.u;
}

// W[K][N] f32 -> WT[N][K] f16 (32x32 LDS tiles)
__global__ __launch_bounds__(256) void k_trans(const float* __restrict__ W,
                                               _Float16* __restrict__ WT,
                                               int K, int N) {
  __shared__ float ts[32][33];
  const int k0 = blockIdx.x * 32, n0 = blockIdx.y * 32;
  const int c = threadIdx.x & 31, r8 = threadIdx.x >> 5;
#pragma unroll
  for (int i = 0; i < 4; ++i) {
    const int r = r8 + i * 8;
    ts[r][c] = W[(size_t)(k0 + r) * N + n0 + c];
  }
  __syncthreads();
#pragma unroll
  for (int i = 0; i < 4; ++i) {
    const int r = r8 + i * 8;
    WT[(size_t)(n0 + r) * K + k0 + c] = (_Float16)ts[c][r];
  }
}

// ---------------- 1) S4D discretization params ------------------------------

__global__ __launch_bounds__(256) void k_params(
    const float* __restrict__ log_dt, const float* __restrict__ Cp,
    const float* __restrict__ log_A_real, const float* __restrict__ A_imag,
    float4* __restrict__ P) {
  const int idx = blockIdx.x * 256 + threadIdx.x;   // < 512*32
  const int h = idx >> 5;
  const float dt = expf(log_dt[h]);
  const float Ar = -expf(log_A_real[idx]);
  const float Ai = A_imag[idx];
  const float ew = expf(Ar * dt);
  const float wr = ew * cosf(Ai * dt);
  const float wi = ew * sinf(Ai * dt);
  const float er = wr - 1.0f, ei = wi;
  const float den = 1.0f / (Ar * Ar + Ai * Ai);
  const float qr = (er * Ar + ei * Ai) * den;
  const float qi = (ei * Ar - er * Ai) * den;
  const float Cr = Cp[2 * idx], Ci = Cp[2 * idx + 1];
  P[idx] = make_float4(wr, wi, 2.0f * (Cr * qr - Ci * qi),
                               2.0f * (Cr * qi + Ci * qr));
}

// ---------------- 2) in_proj GEMM (f16 in, f32 out, masked) -----------------
// 128x128 block tile, 8 waves (2x4), 64x32 per wave, double-buffered LDS.

__global__ __launch_bounds__(256) void k_gemm_inproj(
    const _Float16* __restrict__ Ah, const _Float16* __restrict__ BT,
    const float* __restrict__ bias, const unsigned char* __restrict__ Mk,
    float* __restrict__ X) {
  __shared__ _Float16 As[2][128 * LSTR];
  __shared__ _Float16 Bs[2][128 * LSTR];
  const int K = DIN, NK = DIN / 32;
  const int n0 = blockIdx.x * 128, row0 = blockIdx.y * 128;
  const int t = threadIdx.x, lane = t & 31, wid = t >> 5;
  const int wr_ = wid >> 2, wc_ = wid & 3;
  v8f acc[4][2];
  const v8f vz = {0.f, 0.f, 0.f, 0.f, 0.f, 0.f, 0.f, 0.f};
#pragma unroll
  for (int i = 0; i < 4; ++i)
#pragma unroll
    for (int j = 0; j < 2; ++j) acc[i][j] = vz;

  stage_t(Ah, row0, 0, K, As[0], t);
  stage_t(BT, n0, 0, K, Bs[0], t);
  for (int kt = 0; kt < NK; ++kt) {
    const int cur = kt & 1;
    if (kt + 1 < NK) {
      stage_t(Ah, row0, (kt + 1) * 32, K, As[cur ^ 1], t);
      stage_t(BT, n0, (kt + 1) * 32, K, Bs[cur ^ 1], t);
    }
    WAIT_ASYNC(4);      // this wave's tile-kt copies complete
    __syncthreads();    // => all waves' tile-kt copies complete
    v16h af[4], bf[2];
#pragma unroll
    for (int tm = 0; tm < 4; ++tm) af[tm] = frag_A(As[cur], wr_ * 64 + tm * 16, lane);
#pragma unroll
    for (int tn = 0; tn < 2; ++tn) bf[tn] = frag_B(Bs[cur], wc_ * 32 + tn * 16, lane);
#pragma unroll
    for (int tm = 0; tm < 4; ++tm)
#pragma unroll
      for (int tn = 0; tn < 2; ++tn)
        acc[tm][tn] = wmma_f16(af[tm], bf[tn], acc[tm][tn]);
    __syncthreads();    // cur buffer free for overwrite next iteration
  }
  const int half = lane >> 4, nn = lane & 15;
#pragma unroll
  for (int tm = 0; tm < 4; ++tm)
#pragma unroll
    for (int tn = 0; tn < 2; ++tn) {
      const int gc = n0 + wc_ * 32 + tn * 16 + nn;
      const float bv = bias[gc];
#pragma unroll
      for (int r = 0; r < 8; ++r) {
        const int gr = row0 + wr_ * 64 + tm * 16 + r + 8 * half;
        float v = acc[tm][tn][r] + bv;
        v *= Mk[gr] ? 1.0f : 0.0f;
        X[(size_t)gr * H_DIM + gc] = v;
      }
    }
}

// ---------------- 3) diagonal-SSM scan + skip + GELU -> G (b,t,h) f16 -------
// One wave per (b,h); lane = state. LDS-transposed 8h x 32t output chunks.

__global__ __launch_bounds__(256) void k_scan(
    const float* __restrict__ X, const float4* __restrict__ P,
    const float* __restrict__ Dp, _Float16* __restrict__ Gt) {
  __shared__ _Float16 Gs[8][40];
  const int wid = threadIdx.x >> 5, lane = threadIdx.x & 31;
  const int gid = blockIdx.x * 8 + wid;
  const int b = gid >> 9;           // / H_DIM
  const int h = gid & (H_DIM - 1);
  const int h0 = h & ~7;
  const float4 p = P[h * 32 + lane];
  const float wr = p.x, wi = p.y, cr = p.z, ci = p.w;
  const float Dh = Dp[h];
  float sr = 0.f, si = 0.f;
  const size_t xrow = (size_t)b * T_LEN;
  const int tt = threadIdx.x >> 3, hh = threadIdx.x & 7;
  for (int t0 = 0; t0 < T_LEN; t0 += 32) {
    const float ul = X[(xrow + t0 + lane) * H_DIM + h];
    if (t0 + 32 < T_LEN)
      __builtin_prefetch(&X[(xrow + t0 + 32 + lane) * H_DIM + h], 0, 0);
    float ybuf = 0.f;
#pragma unroll 4
    for (int j = 0; j < 32; ++j) {
      const float u = __shfl(ul, j, 32);
      const float nsr = fmaf(wr, sr, fmaf(-wi, si, u));
      const float nsi = fmaf(wr, si, wi * sr);
      sr = nsr; si = nsi;
      float r = fmaf(cr, sr, -ci * si);
#pragma unroll
      for (int off = 16; off > 0; off >>= 1) r += __shfl_xor(r, off, 32);
      if (lane == j) ybuf = r;
    }
    const float yv = fmaf(Dh, ul, ybuf);
    const float gl = 0.5f * yv * (1.0f + erff(yv * 0.70710678118654752f));
    Gs[wid][lane] = (_Float16)gl;
    __syncthreads();
    Gt[((size_t)(b * T_LEN + t0 + tt)) * H_DIM + h0 + hh] = Gs[hh][tt];
    __syncthreads();
  }
}

// ---------------- 4) GLU GEMM: dual column tiles (o, o+H), f16 out ----------

__global__ __launch_bounds__(256) void k_gemm_glu(
    const _Float16* __restrict__ Gh, const _Float16* __restrict__ WTh,
    const float* __restrict__ GB, _Float16* __restrict__ Y) {
  __shared__ _Float16 As[2][128 * LSTR];
  __shared__ _Float16 Bs[2][2][128 * LSTR];
  const int K = H_DIM, NK = H_DIM / 32;
  const int n0 = blockIdx.x * 128, row0 = blockIdx.y * 128;
  const int t = threadIdx.x, lane = t & 31, wid = t >> 5;
  const int wr_ = wid >> 2, wc_ = wid & 3;
  v8f acca[4][2], accb[4][2];
  const v8f vz = {0.f, 0.f, 0.f, 0.f, 0.f, 0.f, 0.f, 0.f};
#pragma unroll
  for (int i = 0; i < 4; ++i)
#pragma unroll
    for (int j = 0; j < 2; ++j) { acca[i][j] = vz; accb[i][j] = vz; }

  stage_t(Gh, row0, 0, K, As[0], t);
  stage_t(WTh, n0, 0, K, Bs[0][0], t);
  stage_t(WTh, n0 + H_DIM, 0, K, Bs[0][1], t);
  for (int kt = 0; kt < NK; ++kt) {
    const int cur = kt & 1;
    if (kt + 1 < NK) {
      stage_t(Gh, row0, (kt + 1) * 32, K, As[cur ^ 1], t);
      stage_t(WTh, n0, (kt + 1) * 32, K, Bs[cur ^ 1][0], t);
      stage_t(WTh, n0 + H_DIM, (kt + 1) * 32, K, Bs[cur ^ 1][1], t);
    }
    WAIT_ASYNC(6);
    __syncthreads();
    v16h af[4], bfa[2], bfb[2];
#pragma unroll
    for (int tm = 0; tm < 4; ++tm) af[tm] = frag_A(As[cur], wr_ * 64 + tm * 16, lane);
#pragma unroll
    for (int tn = 0; tn < 2; ++tn) {
      bfa[tn] = frag_B(Bs[cur][0], wc_ * 32 + tn * 16, lane);
      bfb[tn] = frag_B(Bs[cur][1], wc_ * 32 + tn * 16, lane);
    }
#pragma unroll
    for (int tm = 0; tm < 4; ++tm)
#pragma unroll
      for (int tn = 0; tn < 2; ++tn) {
        acca[tm][tn] = wmma_f16(af[tm], bfa[tn], acca[tm][tn]);
        accb[tm][tn] = wmma_f16(af[tm], bfb[tn], accb[tm][tn]);
      }
    __syncthreads();
  }
  const int half = lane >> 4, nn = lane & 15;
#pragma unroll
  for (int tm = 0; tm < 4; ++tm)
#pragma unroll
    for (int tn = 0; tn < 2; ++tn) {
      const int gc = n0 + wc_ * 32 + tn * 16 + nn;
      const float ba_ = GB[gc], bb_ = GB[gc + H_DIM];
#pragma unroll
      for (int r = 0; r < 8; ++r) {
        const int gr = row0 + wr_ * 64 + tm * 16 + r + 8 * half;
        const float av = acca[tm][tn][r] + ba_;
        const float bv = accb[tm][tn][r] + bb_;
        const float sg = 1.0f / (1.0f + expf(-bv));
        Y[(size_t)gr * H_DIM + gc] = (_Float16)(av * sg);
      }
    }
}

// ---------------- 5) out_proj GEMM + residual (f32 out) ---------------------

__global__ __launch_bounds__(256) void k_gemm_out(
    const _Float16* __restrict__ Yh, const _Float16* __restrict__ BT,
    const float* __restrict__ bias, const float* __restrict__ X,
    float* __restrict__ OUT) {
  __shared__ _Float16 As[2][128 * LSTR];
  __shared__ _Float16 Bs[2][128 * LSTR];
  const int K = H_DIM, NK = H_DIM / 32;
  const int n0 = blockIdx.x * 128, row0 = blockIdx.y * 128;
  const int t = threadIdx.x, lane = t & 31, wid = t >> 5;
  const int wr_ = wid >> 2, wc_ = wid & 3;
  v8f acc[4][2];
  const v8f vz = {0.f, 0.f, 0.f, 0.f, 0.f, 0.f, 0.f, 0.f};
#pragma unroll
  for (int i = 0; i < 4; ++i)
#pragma unroll
    for (int j = 0; j < 2; ++j) acc[i][j] = vz;

  stage_t(Yh, row0, 0, K, As[0], t);
  stage_t(BT, n0, 0, K, Bs[0], t);
  for (int kt = 0; kt < NK; ++kt) {
    const int cur = kt & 1;
    if (kt + 1 < NK) {
      stage_t(Yh, row0, (kt + 1) * 32, K, As[cur ^ 1], t);
      stage_t(BT, n0, (kt + 1) * 32, K, Bs[cur ^ 1], t);
    }
    WAIT_ASYNC(4);
    __syncthreads();
    v16h af[4], bf[2];
#pragma unroll
    for (int tm = 0; tm < 4; ++tm) af[tm] = frag_A(As[cur], wr_ * 64 + tm * 16, lane);
#pragma unroll
    for (int tn = 0; tn < 2; ++tn) bf[tn] = frag_B(Bs[cur], wc_ * 32 + tn * 16, lane);
#pragma unroll
    for (int tm = 0; tm < 4; ++tm)
#pragma unroll
      for (int tn = 0; tn < 2; ++tn)
        acc[tm][tn] = wmma_f16(af[tm], bf[tn], acc[tm][tn]);
    __syncthreads();
  }
  const int half = lane >> 4, nn = lane & 15;
#pragma unroll
  for (int tm = 0; tm < 4; ++tm)
#pragma unroll
    for (int tn = 0; tn < 2; ++tn) {
      const int gc = n0 + wc_ * 32 + tn * 16 + nn;
      const float bv = bias[gc];
#pragma unroll
      for (int r = 0; r < 8; ++r) {
        const int gr = row0 + wr_ * 64 + tm * 16 + r + 8 * half;
        OUT[(size_t)gr * H_DIM + gc] =
            acc[tm][tn][r] + bv + X[(size_t)gr * H_DIM + gc];
      }
    }
}

// ---------------- 6) LayerNorm (in-place) + mask ----------------------------

__global__ __launch_bounds__(256) void k_ln(
    float* __restrict__ OUT, const float* __restrict__ g,
    const float* __restrict__ bta, const unsigned char* __restrict__ Mk) {
  const int row = blockIdx.x * 8 + (threadIdx.x >> 5);
  const int lane = threadIdx.x & 31;
  float vals[16];
  float s = 0.f, sq = 0.f;
#pragma unroll
  for (int i = 0; i < 16; ++i) {
    const float v = OUT[(size_t)row * H_DIM + lane + i * 32];
    vals[i] = v; s += v; sq += v * v;
  }
#pragma unroll
  for (int off = 16; off > 0; off >>= 1) {
    s += __shfl_xor(s, off, 32);
    sq += __shfl_xor(sq, off, 32);
  }
  const float mu = s * (1.0f / H_DIM);
  const float var = sq * (1.0f / H_DIM) - mu * mu;
  const float rstd = rsqrtf(var + 1e-5f);
  const float mf = Mk[row] ? 1.0f : 0.0f;
#pragma unroll
  for (int i = 0; i < 16; ++i) {
    const int c = lane + i * 32;
    OUT[(size_t)row * H_DIM + c] = ((vals[i] - mu) * rstd * g[c] + bta[c]) * mf;
  }
}

// ---------------- launch ----------------------------------------------------

extern "C" void kernel_launch(void* const* d_in, const int* in_sizes, int n_in,
                              void* d_out, int out_size, void* d_ws,
                              size_t ws_size, hipStream_t stream) {
  const float* z          = (const float*)d_in[0];
  const unsigned char* mk = (const unsigned char*)d_in[1];
  const float* in_w       = (const float*)d_in[2];
  const float* in_b       = (const float*)d_in[3];
  const float* log_dt     = (const float*)d_in[4];
  const float* Cp         = (const float*)d_in[5];
  const float* log_A_real = (const float*)d_in[6];
  const float* A_imag     = (const float*)d_in[7];
  const float* Dp         = (const float*)d_in[8];
  const float* glu_w      = (const float*)d_in[9];
  const float* glu_b      = (const float*)d_in[10];
  const float* out_w      = (const float*)d_in[11];
  const float* out_b      = (const float*)d_in[12];
  const float* ln_g       = (const float*)d_in[13];
  const float* ln_b       = (const float*)d_in[14];
  float* out = (float*)d_out;

  // workspace layout (bytes, all 16B aligned)
  char* w = (char*)d_ws;
  float4*   P      = (float4*)(w);                       // 256 KB
  _Float16* in_wT  = (_Float16*)(w + 262144);            // 2 MB   [512][2048]
  _Float16* glu_wh = (_Float16*)(w + 2359296);           // 1 MB   [1024][512]
  _Float16* out_wT = (_Float16*)(w + 3407872);           // 0.5 MB [512][512]
  float*    X      = (float*)(w + 3932160);              // 33.5 MB (b,t,h) f32
  _Float16* zh     = (_Float16*)(w + 37486592);          // 67 MB  (b,t,din) f16
  _Float16* Gt     = (_Float16*)(w + 37486592);          // reuse: 16.8 MB (b,t,h)
  _Float16* Y      = (_Float16*)(w + 54263808);          // 16.8 MB (b,t,h)

  k_params<<<64, 256, 0, stream>>>(log_dt, Cp, log_A_real, A_imag, P);
  k_cvt<<<16384, 256, 0, stream>>>(z, zh);                       // z -> f16
  k_cvt<<<256, 256, 0, stream>>>(glu_w, glu_wh);                 // [o][h] = [n][k]
  k_trans<<<dim3(64, 16), 256, 0, stream>>>(in_w, in_wT, DIN, H_DIM);
  k_trans<<<dim3(16, 16), 256, 0, stream>>>(out_w, out_wT, H_DIM, H_DIM);

  k_gemm_inproj<<<dim3(4, 128), 256, 0, stream>>>(zh, in_wT, in_b, mk, X);
  k_scan<<<512, 256, 0, stream>>>(X, P, Dp, Gt);
  k_gemm_glu<<<dim3(4, 128), 256, 0, stream>>>(Gt, glu_wh, glu_b, Y);
  k_gemm_out<<<dim3(4, 128), 256, 0, stream>>>(Y, out_wT, out_b, X, out);
  k_ln<<<2048, 256, 0, stream>>>(out, ln_g, ln_b, mk);
}
